// MultiHeadAttention_43499428774106
// MI455X (gfx1250) — matrix operand order repaired and checked by reading, not compiled, not run on hardware
//
#include <hip/hip_runtime.h>

typedef __bf16 bf16;
typedef __attribute__((ext_vector_type(16))) __bf16 v16bf;
typedef __attribute__((ext_vector_type(8)))  __bf16 v8bf;
typedef __attribute__((ext_vector_type(8)))  float  v8f;

#define B_   2
#define S_   2048
#define H_   16
#define DK_  64
#define DM_  1024
#define MTOK (B_ * S_)   // 4096 tokens

static __device__ __forceinline__ v8f vzero8() {
  v8f z;
#pragma unroll
  for (int i = 0; i < 8; i++) z[i] = 0.0f;
  return z;
}

static __device__ __forceinline__ v16bf cat8(v8bf lo, v8bf hi) {
  return __builtin_shufflevector(lo, hi, 0,1,2,3,4,5,6,7,8,9,10,11,12,13,14,15);
}

static __device__ __forceinline__ v8f wmma_bf16(v16bf a, v16bf b, v8f c) {
  // D = A(16x32 bf16) x B(32x16 bf16) + C(16x16 f32)
  return __builtin_amdgcn_wmma_f32_16x16x32_bf16(false, a, false, b, (short)0, c, false, false);
}

// ---------------------------------------------------------------------------
// CDNA5 async global->LDS copy (GLOBAL_LOAD_ASYNC_TO_LDS_B128, ASYNCcnt).
// vdst VGPR carries the LDS byte offset (generic LDS pointer low 32 bits),
// vaddr is the 64-bit global address, SADDR=off.
static __device__ __forceinline__ void async_ld_b128(void* ldsDst, const void* gsrc) {
  uint32_t l = (uint32_t)(uintptr_t)ldsDst;   // low 32 bits = LDS byte offset
  asm volatile("global_load_async_to_lds_b128 %0, %1, off"
               :: "v"(l), "v"(gsrc) : "memory");
}
static __device__ __forceinline__ void async_wait0() {
  asm volatile("s_wait_asynccnt 0x0" ::: "memory");
}

// A-fragment (16x32 MxK) from LDS tile, row stride `ld` elements.
// ISA layout: lanes 0-15: M=lane, elems 0..7 = K0..7, 8..15 = K16..23
//             lanes 16-31: M=lane-16, elems 0..7 = K8..15, 8..15 = K24..31
static __device__ __forceinline__ v16bf fragA(const bf16* base, int ld, int m0, int k0,
                                              int lane16, int laneHalf) {
  const bf16* p = base + (size_t)(m0 + lane16) * ld + k0 + (laneHalf ? 8 : 0);
  v8bf lo = *(const v8bf*)p;
  v8bf hi = *(const v8bf*)(p + 16);
  return cat8(lo, hi);
}

// B-fragment (32x16 KxN) from LDS tile stored N-major [n][k], row stride `ld`.
// ISA layout: lanes 0-15: N=lane, elems 0..7 = K0..7, 8..15 = K16..23; lanes 16-31: N=lane-16, elems 0..7 = K8..15, 8..15 = K24..31
static __device__ __forceinline__ v16bf fragB(const bf16* base, int ld, int n0, int k0,
                                              int lane16, int laneHalf) {
  const bf16* p = base + (size_t)(n0 + lane16) * ld + k0 + (laneHalf ? 8 : 0);
  v8bf lo = *(const v8bf*)p;
  v8bf hi = *(const v8bf*)(p + 16);
  return cat8(lo, hi);
}

// ---------------------------------------------------------------------------
// fp32 -> bf16 convert
__global__ __launch_bounds__(256) void cvt_f32_bf16(const float* __restrict__ src,
                                                    bf16* __restrict__ dst, int n) {
  int i = blockIdx.x * 256 + threadIdx.x;
  if (i < n) dst[i] = (bf16)src[i];
}

// W [K,N] fp32 -> WT [N,K] bf16 (dst index fastest along K -> coalesced stores)
__global__ __launch_bounds__(256) void cvt_w_transpose(const float* __restrict__ src,
                                                       bf16* __restrict__ dst,
                                                       int K, int N) {
  int i = blockIdx.x * 256 + threadIdx.x;
  if (i < K * N) {
    int k = i % K, n = i / K;
    dst[i] = (bf16)src[(size_t)k * N + n];
  }
}

// ---------------------------------------------------------------------------
// bf16 GEMM, double-buffered LDS with async global->LDS staging.
//   A  [M,K] row-major bf16
//   BT [N,K] row-major bf16 (transposed weight)
//   mode 1: out bf16 head-split  [B,H,S,DK]  (for Q,K)
//   mode 2: out bf16 head-split  [B,H,DK,S]  (for V, pre-transposed for P@V)
//   mode 3: out f32 row-major    [M,N]       (final projection -> d_out)
//   else  : out bf16 row-major   [M,N]
__global__ __launch_bounds__(256) void gemm_bf16(
    const bf16* __restrict__ A, const bf16* __restrict__ BT,
    const float* __restrict__ bias,
    bf16* __restrict__ outB, float* __restrict__ outF,
    int M, int N, int K, int mode) {
  __shared__ __align__(128) bf16 aT[2][128][32];
  __shared__ __align__(128) bf16 bT[2][128][32];

  const int tid = threadIdx.x;
  const int lane = tid & 31, wave = tid >> 5;
  const int lane16 = lane & 15, laneHalf = lane >> 4;
  const int waveM = wave >> 2;   // 0..1 -> 64 rows
  const int waveN = wave & 3;    // 0..3 -> 32 cols
  const int m0 = blockIdx.y * 128;
  const int n0 = blockIdx.x * 128;

  const int cpRow = tid >> 1;         // 0..127
  const int cpK   = (tid & 1) * 16;   // 0 or 16

  v8f acc[4][2];
#pragma unroll
  for (int i = 0; i < 4; i++)
#pragma unroll
    for (int j = 0; j < 2; j++) acc[i][j] = vzero8();

  const bf16* aSrc = A  + (size_t)(m0 + cpRow) * K + cpK;
  const bf16* bSrc = BT + (size_t)(n0 + cpRow) * K + cpK;

  // prologue: stage K-tile 0 into buffer 0 (each thread: 2x16B per tile)
  async_ld_b128(&aT[0][cpRow][cpK],     aSrc);
  async_ld_b128(&aT[0][cpRow][cpK + 8], aSrc + 8);
  async_ld_b128(&bT[0][cpRow][cpK],     bSrc);
  async_ld_b128(&bT[0][cpRow][cpK + 8], bSrc + 8);

  int buf = 0;
  for (int k0 = 0; k0 < K; k0 += 32) {
    async_wait0();        // this wave's async writes into `buf` are done
    __syncthreads();      // everyone's writes visible

    if (k0 + 32 < K) {    // kick off next tile into the other buffer
      const bf16* aN = aSrc + k0 + 32;
      const bf16* bN = bSrc + k0 + 32;
      async_ld_b128(&aT[buf ^ 1][cpRow][cpK],     aN);
      async_ld_b128(&aT[buf ^ 1][cpRow][cpK + 8], aN + 8);
      async_ld_b128(&bT[buf ^ 1][cpRow][cpK],     bN);
      async_ld_b128(&bT[buf ^ 1][cpRow][cpK + 8], bN + 8);
    }

    v16bf bFr[2];
#pragma unroll
    for (int ni = 0; ni < 2; ni++)
      bFr[ni] = fragB(&bT[buf][0][0], 32, waveN * 32 + ni * 16, 0, lane16, laneHalf);
#pragma unroll
    for (int mi = 0; mi < 4; mi++) {
      v16bf aFr = fragA(&aT[buf][0][0], 32, waveM * 64 + mi * 16, 0, lane16, laneHalf);
#pragma unroll
      for (int ni = 0; ni < 2; ni++)
        acc[mi][ni] = wmma_bf16(aFr, bFr[ni], acc[mi][ni]);
    }
    buf ^= 1;
  }

  // Epilogue: C/D layout -> lane16 = column, laneHalf*8 + r = row within 16x16
#pragma unroll
  for (int mi = 0; mi < 4; mi++) {
#pragma unroll
    for (int ni = 0; ni < 2; ni++) {
      const int col = n0 + waveN * 32 + ni * 16 + lane16;
      const float bvv = bias[col];
#pragma unroll
      for (int r = 0; r < 8; r++) {
        const int row = m0 + waveM * 64 + mi * 16 + laneHalf * 8 + r;
        float v = acc[mi][ni][r] + bvv;
        if (mode == 3) {
          outF[(size_t)row * N + col] = v;
        } else if (mode == 1) {
          int b = row >> 11, s = row & (S_ - 1), h = col >> 6, d = col & (DK_ - 1);
          outB[(((size_t)(b * H_ + h)) * S_ + s) * DK_ + d] = (bf16)v;
        } else if (mode == 2) {
          int b = row >> 11, s = row & (S_ - 1), h = col >> 6, d = col & (DK_ - 1);
          outB[(((size_t)(b * H_ + h)) * DK_ + d) * S_ + s] = (bf16)v;
        } else {
          outB[(size_t)row * N + col] = (bf16)v;
        }
      }
    }
  }
}

// ---------------------------------------------------------------------------
// Causal flash attention: 64-query block per workgroup, 4 waves, WMMA QK^T and PV.
//   Qh,Kh [B,H,S,DK] bf16 ; Vt [B,H,DK,S] bf16 ; ctx [B,S,H*DK] bf16
__global__ __launch_bounds__(128) void attn_kernel(
    const bf16* __restrict__ Qh, const bf16* __restrict__ Kh,
    const bf16* __restrict__ Vt, bf16* __restrict__ ctx) {
  __shared__ __align__(128) bf16 qT[64][64];
  __shared__ __align__(128) bf16 kT[64][64];
  __shared__ __align__(128) bf16 vT[64][64];   // [d][kv]
  __shared__ __align__(128) float sT[64][64];
  __shared__ __align__(128) bf16 pT[64][64];
  __shared__ float mRow[64], lRow[64], scRow[64];

  const int qb = blockIdx.x;   // query block 0..31
  const int bh = blockIdx.y;   // (b*H + h)    0..31
  const int tid = threadIdx.x;
  const int lane = tid & 31, wave = tid >> 5;
  const int lane16 = lane & 15, laneHalf = lane >> 4;
  const int wq0 = wave * 16;   // this wave's query-row slice

  const bf16* Qb = Qh + (size_t)bh * S_ * DK_;
  const bf16* Kb = Kh + (size_t)bh * S_ * DK_;
  const bf16* Vb = Vt + (size_t)bh * DK_ * S_;

  const int cr = tid >> 1, cc = (tid & 1) * 32;   // staging slot (64B / thread)

  { // stage Q block (64x64 bf16) via async copy
    const bf16* s = Qb + (size_t)(qb * 64 + cr) * DK_ + cc;
    async_ld_b128(&qT[cr][cc],      s);
    async_ld_b128(&qT[cr][cc + 8],  s + 8);
    async_ld_b128(&qT[cr][cc + 16], s + 16);
    async_ld_b128(&qT[cr][cc + 24], s + 24);
  }
  if (tid < 64) { mRow[tid] = -3.0e38f; lRow[tid] = 0.0f; }

  v8f oAcc[4];
#pragma unroll
  for (int i = 0; i < 4; i++) oAcc[i] = vzero8();

  for (int kb = 0; kb <= qb; kb++) {   // causal: only blocks up to the diagonal
    __syncthreads();                   // previous iteration's kT/vT/pT reads done
    { // stage K block and V^T block via async copy
      const bf16* sk = Kb + (size_t)(kb * 64 + cr) * DK_ + cc;
      async_ld_b128(&kT[cr][cc],      sk);
      async_ld_b128(&kT[cr][cc + 8],  sk + 8);
      async_ld_b128(&kT[cr][cc + 16], sk + 16);
      async_ld_b128(&kT[cr][cc + 24], sk + 24);
      const bf16* sv = Vb + (size_t)cr * S_ + kb * 64 + cc;
      async_ld_b128(&vT[cr][cc],      sv);
      async_ld_b128(&vT[cr][cc + 8],  sv + 8);
      async_ld_b128(&vT[cr][cc + 16], sv + 16);
      async_ld_b128(&vT[cr][cc + 24], sv + 24);
    }
    async_wait0();
    __syncthreads();

    // S = Q K^T : rows [wq0, wq0+16), all 64 kv columns
    v16bf aQ[2];
#pragma unroll
    for (int ks = 0; ks < 2; ks++)
      aQ[ks] = fragA(&qT[0][0], 64, wq0, ks * 32, lane16, laneHalf);
#pragma unroll
    for (int ni = 0; ni < 4; ni++) {
      v8f sAcc = vzero8();
#pragma unroll
      for (int ks = 0; ks < 2; ks++) {
        v16bf bK = fragB(&kT[0][0], 64, ni * 16, ks * 32, lane16, laneHalf);
        sAcc = wmma_bf16(aQ[ks], bK, sAcc);
      }
#pragma unroll
      for (int r = 0; r < 8; r++) {
        const int lq = wq0 + laneHalf * 8 + r;
        const int lc = ni * 16 + lane16;
        float v = sAcc[r] * 0.125f;                    // 1/sqrt(DK)
        if (kb * 64 + lc > qb * 64 + lq) v = -1.0e9f;  // causal mask
        sT[lq][lc] = v;
      }
    }
    __syncthreads();

    // online softmax: one row per thread
    if (tid < 64) {
      float mOld = mRow[tid];
      float rmax = -3.0e38f;
      for (int c = 0; c < 64; c++) rmax = fmaxf(rmax, sT[tid][c]);
      float mNew = fmaxf(mOld, rmax);
      float sc = __expf(mOld - mNew);
      float sum = 0.0f;
      for (int c = 0; c < 64; c++) {
        float p = __expf(sT[tid][c] - mNew);
        pT[tid][c] = (bf16)p;
        sum += p;
      }
      lRow[tid] = lRow[tid] * sc + sum;
      mRow[tid] = mNew;
      scRow[tid] = sc;
    }
    __syncthreads();

    // rescale running O, then O += P V
#pragma unroll
    for (int r = 0; r < 8; r++) {
      const float sc = scRow[wq0 + laneHalf * 8 + r];
#pragma unroll
      for (int di = 0; di < 4; di++) oAcc[di][r] *= sc;
    }
#pragma unroll
    for (int ks = 0; ks < 2; ks++) {
      v16bf aP = fragA(&pT[0][0], 64, wq0, ks * 32, lane16, laneHalf);
#pragma unroll
      for (int di = 0; di < 4; di++) {
        v16bf bV = fragB(&vT[0][0], 64, di * 16, ks * 32, lane16, laneHalf);
        oAcc[di] = wmma_bf16(aP, bV, oAcc[di]);
      }
    }
  }
  __syncthreads();

  // normalize and write ctx [B,S,H*DK]
  const int b = bh >> 4, h = bh & (H_ - 1);
#pragma unroll
  for (int di = 0; di < 4; di++) {
#pragma unroll
    for (int r = 0; r < 8; r++) {
      const int lq = wq0 + laneHalf * 8 + r;
      const int gq = qb * 64 + lq;
      const int d  = di * 16 + lane16;
      float v = oAcc[di][r] / lRow[lq];
      ctx[((size_t)(b * S_ + gq)) * DM_ + h * DK_ + d] = (bf16)v;
    }
  }
}

// ---------------------------------------------------------------------------
extern "C" void kernel_launch(void* const* d_in, const int* in_sizes, int n_in,
                              void* d_out, int out_size, void* d_ws, size_t ws_size,
                              hipStream_t stream) {
  (void)in_sizes; (void)n_in; (void)out_size; (void)ws_size;
  const float* query = (const float*)d_in[0];
  const float* key_i = (const float*)d_in[1];
  const float* value = (const float*)d_in[2];
  // d_in[3]: causal mask, known analytically -> unused
  const float* Wq = (const float*)d_in[4];
  const float* bq = (const float*)d_in[5];
  const float* Wk = (const float*)d_in[6];
  const float* bk = (const float*)d_in[7];
  const float* Wv = (const float*)d_in[8];
  const float* bv = (const float*)d_in[9];
  const float* Wo = (const float*)d_in[10];
  const float* bo = (const float*)d_in[11];

  char* ws = (char*)d_ws;
  size_t off = 0;
  auto take = [&](size_t bytes) -> void* {
    void* p = ws + off;
    off += (bytes + 255) & ~(size_t)255;
    return p;
  };
  const size_t tokBytes = (size_t)MTOK * DM_ * sizeof(bf16);   // 8 MB
  const size_t wBytes   = (size_t)DM_ * DM_ * sizeof(bf16);    // 2 MB
  bf16* qin = (bf16*)take(tokBytes);
  bf16* kin = (bf16*)take(tokBytes);
  bf16* vin = (bf16*)take(tokBytes);
  bf16* WqT = (bf16*)take(wBytes);
  bf16* WkT = (bf16*)take(wBytes);
  bf16* WvT = (bf16*)take(wBytes);
  bf16* WoT = (bf16*)take(wBytes);
  bf16* Qh  = (bf16*)take(tokBytes);
  bf16* Kh  = (bf16*)take(tokBytes);
  bf16* Vt  = (bf16*)take(tokBytes);
  bf16* ctx = (bf16*)take(tokBytes);   // total ~64 MB

  const int nTok = MTOK * DM_;
  cvt_f32_bf16<<<(nTok + 255) / 256, 256, 0, stream>>>(query, qin, nTok);
  cvt_f32_bf16<<<(nTok + 255) / 256, 256, 0, stream>>>(key_i, kin, nTok);
  cvt_f32_bf16<<<(nTok + 255) / 256, 256, 0, stream>>>(value, vin, nTok);
  const int nW = DM_ * DM_;
  cvt_w_transpose<<<(nW + 255) / 256, 256, 0, stream>>>(Wq, WqT, DM_, DM_);
  cvt_w_transpose<<<(nW + 255) / 256, 256, 0, stream>>>(Wk, WkT, DM_, DM_);
  cvt_w_transpose<<<(nW + 255) / 256, 256, 0, stream>>>(Wv, WvT, DM_, DM_);
  cvt_w_transpose<<<(nW + 255) / 256, 256, 0, stream>>>(Wo, WoT, DM_, DM_);

  dim3 ggrid(DM_ / 128, MTOK / 128);   // (8, 32)
  gemm_bf16<<<ggrid, 256, 0, stream>>>(qin, WqT, bq, Qh, nullptr, MTOK, DM_, DM_, 1);
  gemm_bf16<<<ggrid, 256, 0, stream>>>(kin, WkT, bk, Kh, nullptr, MTOK, DM_, DM_, 1);
  gemm_bf16<<<ggrid, 256, 0, stream>>>(vin, WvT, bv, Vt, nullptr, MTOK, DM_, DM_, 2);

  attn_kernel<<<dim3(S_ / 64, B_ * H_), 128, 0, stream>>>(Qh, Kh, Vt, ctx);

  gemm_bf16<<<ggrid, 256, 0, stream>>>(ctx, WoT, bo, nullptr, (float*)d_out,
                                       MTOK, DM_, DM_, 3);
}
